// MixedDecoder_858993459256
// MI455X (gfx1250) — matrix operand order repaired and checked by reading, not compiled, not run on hardware
//
#include <hip/hip_runtime.h>
#include <math.h>

// ---------------------------------------------------------------------------
// MixedDecoder (MoE) for gfx1250 — fragment-packed bf16 WMMA version.
//   B=512, L=64, FS=96, H=512, E=8, IN0=160, IN12=576, OUT=96
//
//   gating_kernel : f32 VALU MLP -> coeff [512,8]
//   pack_w_kernel : W [E,IN,H] f32 -> PW[kt][nt][e][lane][16] bf16 (B-fragments)
//   pack_a_kernel : concat(z,xb) f32 -> PA[rt][kt][lane][16] bf16 (A-fragments)
//   moe_layer     : per 16x16 tile: depth-2 pipelined K-loop — each WMMA
//                   overlaps the fetch of the B-fragment two experts ahead
//                   (sched_group_barrier pattern, ~110 VGPRs, no spills).
// ---------------------------------------------------------------------------

typedef __attribute__((ext_vector_type(16))) __bf16 v16bf;
typedef __attribute__((ext_vector_type(8)))  float  v8f;

#define NB   512
#define NEXP 8
#define ZC   64

#if defined(__has_builtin)
#  if __has_builtin(__builtin_amdgcn_sched_group_barrier)
#    define SCHED_GROUP(mask, size, id) __builtin_amdgcn_sched_group_barrier(mask, size, id)
#  endif
#endif
#ifndef SCHED_GROUP
#  define SCHED_GROUP(mask, size, id)
#endif

constexpr int KT0  = 160 / 32;  // 5  k-tiles, layer 0
constexpr int KT12 = 576 / 32;  // 18 k-tiles, layers 1/2
constexpr int NT01 = 512 / 16;  // 32 n-tiles, layers 0/1
constexpr int NT2  = 96 / 16;   // 6  n-tiles, layer 2
constexpr int RT   = NB / 16;   // 32 row tiles
constexpr int FRAG = 32 * 16;   // bf16 elements per packed 32x16 tile (512)

__device__ __forceinline__ float elu_f(float x) {
    return x > 0.0f ? x : (__expf(x) - 1.0f);
}

// ---------------------------------------------------------------------------
// Gating network (f32, one thread per row).
// ---------------------------------------------------------------------------
__global__ void gating_kernel(const float* __restrict__ z,   // [512,64]
                              const float* __restrict__ c,   // [512,96]
                              const float* __restrict__ gw0, // [160,64]
                              const float* __restrict__ gb0,
                              const float* __restrict__ gw1, // [64,64]
                              const float* __restrict__ gb1,
                              const float* __restrict__ gw2, // [64,8]
                              const float* __restrict__ gb2,
                              float* __restrict__ coeff)     // [512,8]
{
    const int row = blockIdx.x * blockDim.x + threadIdx.x;
    if (row >= NB) return;

    float g0[64];
    #pragma unroll
    for (int j = 0; j < 64; ++j) g0[j] = gb0[j];
    for (int i = 0; i < 160; ++i) {
        const float xv = (i < ZC) ? z[row * ZC + i] : c[row * 96 + (i - ZC)];
        #pragma unroll
        for (int j = 0; j < 64; ++j) g0[j] += xv * gw0[i * 64 + j];
    }
    #pragma unroll
    for (int j = 0; j < 64; ++j) g0[j] = elu_f(g0[j]);

    float g1[64];
    #pragma unroll
    for (int j = 0; j < 64; ++j) g1[j] = gb1[j];
    for (int i = 0; i < 64; ++i) {
        const float xv = g0[i];
        #pragma unroll
        for (int j = 0; j < 64; ++j) g1[j] += xv * gw1[i * 64 + j];
    }
    #pragma unroll
    for (int j = 0; j < 64; ++j) g1[j] = elu_f(g1[j]);

    float logit[NEXP];
    #pragma unroll
    for (int e = 0; e < NEXP; ++e) logit[e] = gb2[e];
    for (int i = 0; i < 64; ++i) {
        const float xv = g1[i];
        #pragma unroll
        for (int e = 0; e < NEXP; ++e) logit[e] += xv * gw2[i * NEXP + e];
    }
    float mx = logit[0];
    #pragma unroll
    for (int e = 1; e < NEXP; ++e) mx = fmaxf(mx, logit[e]);
    float s = 0.0f, ex[NEXP];
    #pragma unroll
    for (int e = 0; e < NEXP; ++e) { ex[e] = __expf(logit[e] - mx); s += ex[e]; }
    const float inv = 1.0f / s;
    #pragma unroll
    for (int e = 0; e < NEXP; ++e) coeff[row * NEXP + e] = ex[e] * inv;
}

// ---------------------------------------------------------------------------
// Pack weights into WMMA B-fragment layout, bf16.
//   B 32x16 packing used here (contiguous k, same as A): lane = N + 16*half, elem j -> K = 16*half + j
//   PW index: (((kt*NT + nt)*NEXP + e)*32 + lane)*16
// grid = (NT, KT, E), block = 32.
// ---------------------------------------------------------------------------
__global__ void pack_w_kernel(const float* __restrict__ W,  // [E, IN, H]
                              __bf16* __restrict__ pw,
                              const int IN, const int H, const int NT)
{
    const int l = threadIdx.x, half = l >> 4, nl = l & 15;
    const int nt = blockIdx.x, kt = blockIdx.y, e = blockIdx.z;
    const int n = nt * 16 + nl;

    v16bf frag;
    #pragma unroll
    for (int j = 0; j < 16; ++j) {
        const int K = kt * 32 + 16 * half + j;
        frag[j] = (__bf16)W[((size_t)e * IN + K) * H + n];
    }
    *(v16bf*)(pw + ((((size_t)kt * NT + nt) * NEXP + e) * 32 + l) * 16) = frag;
}

// ---------------------------------------------------------------------------
// Pack activations concat(z, xb) into WMMA A-fragment layout, bf16.
//   A 16x32 packing used here (contiguous k, same as B): lane = M + 16*half;
//     elem j -> K = 16*half + j
//   PA index: ((rt*KT + kt)*32 + lane)*16
// grid = (KT, RT), block = 32.
// ---------------------------------------------------------------------------
__global__ void pack_a_kernel(const float* __restrict__ z,   // [512,64]
                              const float* __restrict__ xb,  // [512,xb_cols]
                              const int xb_cols,
                              __bf16* __restrict__ pa, const int KT)
{
    const int l = threadIdx.x, half = l >> 4;
    const int kt = blockIdx.x, rt = blockIdx.y;
    const int m = rt * 16 + (l & 15);

    v16bf frag;
    #pragma unroll
    for (int j = 0; j < 16; ++j) {
        const int kk = 16 * half + j;
        const int k  = kt * 32 + kk;
        const float v = (k < ZC) ? z[(size_t)m * ZC + k]
                                 : xb[(size_t)m * xb_cols + (k - ZC)];
        frag[j] = (__bf16)v;
    }
    *(v16bf*)(pa + (((size_t)rt * KT + kt) * 32 + l) * 16) = frag;
}

// ---------------------------------------------------------------------------
// MoE layer on packed fragments. One wave per 16x16 output tile.
// Depth-2 pipeline per K-step via sched groups:
//   [VMEM x6] (A,B0,B1) then 6 x ([WMMA x1][VMEM x2]) then [WMMA x2]
// Each WMMA consumes Bi while Bi+2 is in flight; only ~3 B buffers live at
// once (no spills), and waits become partial instead of loadcnt==0.
// ---------------------------------------------------------------------------
__global__ void moe_layer_kernel(const __bf16* __restrict__ pa,
                                 const __bf16* __restrict__ pw,
                                 const float* __restrict__ bias,  // [E, H]
                                 const float* __restrict__ coeff, // [512,8]
                                 float* __restrict__ out,         // [512,H]
                                 const int KT, const int NT, const int H,
                                 const int apply_elu)
{
    const int l = threadIdx.x, half = l >> 4, nl = l & 15;
    const int nt = blockIdx.x, rt = blockIdx.y;
    const int ncol = nt * 16 + nl;

    const __bf16* paw = pa + ((size_t)rt * KT * 32 + l) * 16;
    const __bf16* pww = pw + (((size_t)nt * NEXP) * 32 + l) * 16;
    const size_t pa_stride = FRAG;                        // per k-step
    const size_t pw_stride = (size_t)NT * NEXP * FRAG;    // per k-step

    v8f acc[NEXP];
    #pragma unroll
    for (int e = 0; e < NEXP; ++e)
        acc[e] = (v8f){0.f, 0.f, 0.f, 0.f, 0.f, 0.f, 0.f, 0.f};

    for (int kt = 0; kt < KT; ++kt) {
        const v16bf a = *(const v16bf*)paw;
        v16bf bfrag[NEXP];
        #pragma unroll
        for (int e = 0; e < NEXP; ++e)
            bfrag[e] = *(const v16bf*)(pww + (size_t)e * FRAG);

        #pragma unroll
        for (int e = 0; e < NEXP; ++e)
            acc[e] = __builtin_amdgcn_wmma_f32_16x16x32_bf16(
                         false, a, false, bfrag[e], (short)0, acc[e],
                         false, false);

        // Depth-2 pipeline: A+B0+B1 up front, then WMMA_i || load B_{i+2}.
        SCHED_GROUP(0x010, 6, 0);           // A(2) + B0(2) + B1(2)
        #pragma unroll
        for (int i = 0; i < 6; ++i) {
            SCHED_GROUP(0x008, 1, 0);       // WMMA e=i
            SCHED_GROUP(0x010, 2, 0);       // load B_{i+2}
        }
        SCHED_GROUP(0x008, 2, 0);           // WMMA e=6,7 (drain)

        paw += pa_stride;
        pww += pw_stride;
    }

    // Epilogue: blend experts with coeff, add coeff-weighted bias, ELU, store.
    float bv[NEXP];
    #pragma unroll
    for (int e = 0; e < NEXP; ++e) bv[e] = bias[e * H + ncol];

    #pragma unroll
    for (int r = 0; r < 8; ++r) {
        const int m = rt * 16 + 8 * half + r;
        const v8f cw = *(const v8f*)(coeff + (size_t)m * NEXP);  // 2x b128
        float s = 0.0f;
        #pragma unroll
        for (int e = 0; e < NEXP; ++e)
            s += cw[e] * (acc[e][r] + bv[e]);
        if (apply_elu) s = elu_f(s);
        out[(size_t)m * H + ncol] = s;
    }
}

// ---------------------------------------------------------------------------
extern "C" void kernel_launch(void* const* d_in, const int* in_sizes, int n_in,
                              void* d_out, int out_size, void* d_ws, size_t ws_size,
                              hipStream_t stream) {
    const float* z   = (const float*)d_in[0];
    const float* c   = (const float*)d_in[1];
    const float* gw0 = (const float*)d_in[2];
    const float* gb0 = (const float*)d_in[3];
    const float* gw1 = (const float*)d_in[4];
    const float* gb1 = (const float*)d_in[5];
    const float* gw2 = (const float*)d_in[6];
    const float* gb2 = (const float*)d_in[7];
    const float* w0  = (const float*)d_in[8];
    const float* b0  = (const float*)d_in[9];
    const float* w1  = (const float*)d_in[10];
    const float* b1  = (const float*)d_in[11];
    const float* w2  = (const float*)d_in[12];
    const float* b2  = (const float*)d_in[13];
    float* out = (float*)d_out;

    // ---- workspace layout (all regions 1KB-aligned, ~10 MB total) --------
    char* wsb = (char*)d_ws;
    size_t off = 0;
    float* coeff = (float*)(wsb + off); off += (size_t)NB * NEXP * 4;
    off = (off + 1023) & ~(size_t)1023;
    float* h0    = (float*)(wsb + off); off += (size_t)NB * 512 * 4;
    float* h1    = (float*)(wsb + off); off += (size_t)NB * 512 * 4;
    __bf16* pa0  = (__bf16*)(wsb + off); off += (size_t)RT * KT0  * FRAG * 2;
    __bf16* pa1  = (__bf16*)(wsb + off); off += (size_t)RT * KT12 * FRAG * 2;
    __bf16* pa2  = (__bf16*)(wsb + off); off += (size_t)RT * KT12 * FRAG * 2;
    __bf16* pw0  = (__bf16*)(wsb + off); off += (size_t)KT0  * NT01 * NEXP * FRAG * 2;
    __bf16* pw1  = (__bf16*)(wsb + off); off += (size_t)KT12 * NT01 * NEXP * FRAG * 2;
    __bf16* pw2  = (__bf16*)(wsb + off); off += (size_t)KT12 * NT2  * NEXP * FRAG * 2;
    (void)off;

    // 1) gating
    gating_kernel<<<dim3(NB / 256), dim3(256), 0, stream>>>(
        z, c, gw0, gb0, gw1, gb1, gw2, gb2, coeff);

    // 2) pack weights (independent of activations)
    pack_w_kernel<<<dim3(NT01, KT0,  NEXP), dim3(32), 0, stream>>>(w0, pw0, 160, 512, NT01);
    pack_w_kernel<<<dim3(NT01, KT12, NEXP), dim3(32), 0, stream>>>(w1, pw1, 576, 512, NT01);
    pack_w_kernel<<<dim3(NT2,  KT12, NEXP), dim3(32), 0, stream>>>(w2, pw2, 576, 96,  NT2);

    // 3) layer 0: x = concat(z,c)
    pack_a_kernel<<<dim3(KT0, RT), dim3(32), 0, stream>>>(z, c, 96, pa0, KT0);
    moe_layer_kernel<<<dim3(NT01, RT), dim3(32), 0, stream>>>(
        pa0, pw0, b0, coeff, h0, KT0, NT01, 512, 1);

    // 4) layer 1: x = concat(z,h0)
    pack_a_kernel<<<dim3(KT12, RT), dim3(32), 0, stream>>>(z, h0, 512, pa1, KT12);
    moe_layer_kernel<<<dim3(NT01, RT), dim3(32), 0, stream>>>(
        pa1, pw1, b1, coeff, h1, KT12, NT01, 512, 1);

    // 5) layer 2: x = concat(z,h1) -> out
    pack_a_kernel<<<dim3(KT12, RT), dim3(32), 0, stream>>>(z, h1, 512, pa2, KT12);
    moe_layer_kernel<<<dim3(NT2, RT), dim3(32), 0, stream>>>(
        pa2, pw2, b2, coeff, out, KT12, NT2, 96, 0);
}
